// CausalSelfAttention_74088185856149
// MI455X (gfx1250) — matrix operand order repaired and checked by reading, not compiled, not run on hardware
//
#include <hip/hip_runtime.h>
#include <hip/hip_bf16.h>
#include <math.h>

// ---------------------------------------------------------------------------
// CDNA5 (gfx1250) causal self-attention: bf16 WMMA + async-to-LDS K staging.
// ---------------------------------------------------------------------------

typedef __attribute__((ext_vector_type(16))) __bf16 v16bf;
typedef __attribute__((ext_vector_type(8)))  __bf16 v8bf;
typedef __attribute__((ext_vector_type(8)))  float  v8f;

#define BB 2
#define TT 2048
#define CC 2048
#define HH 16
#define DD 128
#define MROT 32

static __device__ __forceinline__ v8f wmma_bf16(v16bf a, v16bf b, v8f c) {
  // D = A(16x32 bf16) * B(32x16 bf16) + C(16x16 f32)
  return __builtin_amdgcn_wmma_f32_16x16x32_bf16(
      /*neg_a=*/false, a, /*neg_b=*/false, b,
      /*c_mod=*/(short)0, c, /*reuse_a=*/false, /*reuse_b=*/false);
}

static __device__ __forceinline__ v16bf cat8(v8bf lo, v8bf hi) {
  return __builtin_shufflevector(lo, hi, 0,1,2,3,4,5,6,7,8,9,10,11,12,13,14,15);
}

// A fragment (16x32 bf16) from row-major [*, ld] source.
// Lane layout: M = lane%16; K = k0 + {0..15} (lanes 0-15) or
//                               k0 + {16..31} (lanes 16-31).
static __device__ __forceinline__ v16bf load_a(const __bf16* p, int row, int k0,
                                               int ld, int half) {
  const __bf16* q = p + (size_t)row * ld + k0 + 16 * half;
  v8bf lo = *(const v8bf*)q;
  v8bf hi = *(const v8bf*)(q + 8);
  return cat8(lo, hi);
}

// B fragment (32x16 bf16): lane holds N = lane%16, K = 16*(lane/16)+0..15,
// i.e. 16 contiguous bf16 (32 bytes) at caller-computed address.
static __device__ __forceinline__ v16bf load_b16(const __bf16* p) {
  return *(const v16bf*)p;
}

static __device__ __forceinline__ float rowmax16(float v) {
  v = fmaxf(v, __shfl_xor(v, 1));
  v = fmaxf(v, __shfl_xor(v, 2));
  v = fmaxf(v, __shfl_xor(v, 4));
  v = fmaxf(v, __shfl_xor(v, 8));
  return v;
}
static __device__ __forceinline__ float rowsum16(float v) {
  v += __shfl_xor(v, 1);
  v += __shfl_xor(v, 2);
  v += __shfl_xor(v, 4);
  v += __shfl_xor(v, 8);
  return v;
}

// ---------------------------------------------------------------------------
// Conversions
// ---------------------------------------------------------------------------
__global__ void cvt_bf16(const float* __restrict__ in, __bf16* __restrict__ out,
                         int n) {
  int i = blockIdx.x * blockDim.x + threadIdx.x;
  if (i < n) out[i] = (__bf16)in[i];
}

// in: row-major [rows, cols] fp32 -> out: [cols, rows] bf16 (transposed)
__global__ void cvt_T(const float* __restrict__ in, __bf16* __restrict__ out,
                      int rows, int cols) {
  int i = blockIdx.x * blockDim.x + threadIdx.x;
  if (i < rows * cols) {
    int r = i / cols, c = i - r * cols;
    out[(size_t)c * rows + r] = (__bf16)in[i];
  }
}

// ---------------------------------------------------------------------------
// QKV GEMM: [B*T, C] x [C, 3C] -> scatter into Q[B,H,T,D], K[B,H,T,D],
// Vt[B,H,D,T]. One wave per 32(M) x 64(N) tile (2 A-frags, 8 accumulators:
// each B fragment feeds two WMMAs). 8 waves / block.
// ---------------------------------------------------------------------------
__global__ __launch_bounds__(256) void gemm_qkv(
    const __bf16* __restrict__ xb, const __bf16* __restrict__ waT,
    __bf16* __restrict__ qw, __bf16* __restrict__ kw, __bf16* __restrict__ vt) {
  const int lane = threadIdx.x & 31;
  const int wid  = threadIdx.x >> 5;
  const int half = lane >> 4;
  const int nl   = lane & 15;
  const int task = blockIdx.x * 8 + wid;      // 128 * 96 tasks
  const int mt = task / 96;
  const int ns = task - mt * 96;
  const int m0 = mt * 32;
  const int n0 = ns * 64;

  v8f acc[2][4] = {};
  for (int k0 = 0; k0 < CC; k0 += 32) {
    v16bf a0 = load_a(xb, m0 + nl,      k0, CC, half);
    v16bf a1 = load_a(xb, m0 + 16 + nl, k0, CC, half);
#pragma unroll
    for (int t = 0; t < 4; ++t) {
      v16bf bf = load_b16(waT + (size_t)(n0 + 16 * t + nl) * CC + k0 + 16 * half);
      acc[0][t] = wmma_bf16(a0, bf, acc[0][t]);
      acc[1][t] = wmma_bf16(a1, bf, acc[1][t]);
    }
  }
  // Epilogue scatter. C layout: VGPR j -> row (+ j + 8*half), col = 16t + nl.
#pragma unroll
  for (int sub = 0; sub < 2; ++sub) {
#pragma unroll
    for (int t = 0; t < 4; ++t) {
#pragma unroll
      for (int j = 0; j < 8; ++j) {
        const int g  = m0 + sub * 16 + j + 8 * half;  // global row in [0, B*T)
        const int bb = g >> 11;                       // / T
        const int tq = g & (TT - 1);
        const int c  = n0 + 16 * t + nl;              // [0, 3C)
        const int sect = c >> 11;                     // 0=q 1=k 2=v
        const int cc = c & (CC - 1);
        const int h  = cc >> 7;
        const int dd = cc & (DD - 1);
        const size_t bh = (size_t)(bb * HH + h);
        __bf16 v = (__bf16)acc[sub][t][j];
        if (sect == 0)      qw[(bh * TT + tq) * DD + dd] = v;
        else if (sect == 1) kw[(bh * TT + tq) * DD + dd] = v;
        else                vt[(bh * DD + dd) * TT + tq] = v;
      }
    }
  }
}

// ---------------------------------------------------------------------------
// Rotary on q and k (first 2*MROT dims of each head), fp32 math, bf16 store.
// ---------------------------------------------------------------------------
__global__ void rope_kernel(__bf16* __restrict__ qw, __bf16* __restrict__ kw,
                            const float* __restrict__ freqs,
                            const float* __restrict__ delta) {
  const int n = BB * HH * TT * MROT;        // 2,097,152
  int tid = blockIdx.x * blockDim.x + threadIdx.x;
  __bf16* p = (tid < n) ? qw : kw;
  int r = (tid < n) ? tid : tid - n;
  int i  = r & (MROT - 1);
  int t  = (r >> 5) & (TT - 1);
  int bh = r >> 16;                         // / (MROT*TT)
  size_t base = ((size_t)bh * TT + t) * DD;
  float ang = delta[t] * freqs[i];
  float cs = cosf(ang), sn = sinf(ang);
  float u1 = (float)p[base + i];
  float u2 = (float)p[base + MROT + i];
  p[base + i]        = (__bf16)(u1 * cs - u2 * sn);
  p[base + MROT + i] = (__bf16)(u1 * sn + u2 * cs);
}

// ---------------------------------------------------------------------------
// Async staging of one 32-key x 128-d K tile (8 KB) into LDS.
// 16 x global_load_async_to_lds_b128: each moves 16 B/lane * 32 lanes = 512 B.
// Lane l covers row 2i + (l/16), 16-byte column chunk (l%16).
// ---------------------------------------------------------------------------
static __device__ __forceinline__ void stage_k(const __bf16* __restrict__ K,
                                               int k0, __bf16* buf, int lane) {
  const int half = lane >> 4;
  const int nl   = lane & 15;
  unsigned long long g =
      (unsigned long long)(uintptr_t)(K + (size_t)(k0 + half) * DD + nl * 8);
  // Low 32 bits of a generic shared pointer are the LDS byte offset.
  unsigned l = (unsigned)(uintptr_t)buf + (unsigned)(lane * 16);
#pragma unroll
  for (int i = 0; i < 16; ++i) {
    asm volatile("global_load_async_to_lds_b128 %0, %1, off"
                 :: "v"(l), "v"(g) : "memory");
    g += 2u * DD * sizeof(__bf16);   // advance 2 rows (512 B)
    l += 512u;
  }
}

// ---------------------------------------------------------------------------
// Flash attention: one wave (32 threads) per (b, h, 16-query tile).
// K tiles double-buffered in LDS via async loads (ASYNCcnt); online softmax;
// P transposed C-layout -> A-layout through 1 KB LDS. V streams from L2.
// ---------------------------------------------------------------------------
__global__ __launch_bounds__(32) void attn_kernel(
    const __bf16* __restrict__ qw, const __bf16* __restrict__ kw,
    const __bf16* __restrict__ vt, __bf16* __restrict__ yb) {
  __shared__ __align__(64) __bf16 kbuf[2][32 * DD];   // 16 KB, double buffer
  __shared__ __align__(32) __bf16 plds[16 * 32];      // 1 KB P transpose
  const int lane = threadIdx.x & 31;
  const int half = lane >> 4;
  const int nl   = lane & 15;
  const int task = blockIdx.x;              // B*H*(T/16) = 4096
  const int qt = task & 127;
  const int bh = task >> 7;
  const int bb = bh >> 4;
  const int h  = bh & (HH - 1);
  const int qb = qt * 16;

  const __bf16* Q = qw + (size_t)bh * TT * DD;
  const __bf16* K = kw + (size_t)bh * TT * DD;
  const __bf16* V = vt + (size_t)bh * DD * TT;

  v16bf aQ[4];
#pragma unroll
  for (int c = 0; c < 4; ++c) aQ[c] = load_a(Q, qb + nl, 32 * c, DD, half);

  v8f acc[8] = {};
  float m[8], l[8];
#pragma unroll
  for (int j = 0; j < 8; ++j) { m[j] = -1e30f; l[j] = 0.f; }

  const float scale = 0.08838834764831845f;     // 1/sqrt(128)
  const int nkb = (qb + 47) >> 5;               // ceil((qb+16)/32)

  stage_k(K, 0, kbuf[0], lane);                 // prefetch tile 0
  for (int kb = 0; kb < nkb; ++kb) {
    const int k0 = kb * 32;
    // Stream the next tile (or a harmless dummy on the last iteration) so the
    // outstanding-async bound is uniform; async loads complete in order, so
    // ASYNCcnt <= 16 proves the current tile is fully in LDS.
    const int k0n = (kb + 1 < nkb) ? (kb + 1) * 32 : 0;
    stage_k(K, k0n, kbuf[(kb + 1) & 1], lane);
    asm volatile("s_wait_asynccnt 0x10" ::: "memory");
    const __bf16* kt = kbuf[kb & 1];

    v8f s0 = {}, s1 = {};
#pragma unroll
    for (int c = 0; c < 4; ++c) {
      v16bf b0 = *(const v16bf*)(kt + (size_t)nl * DD + 32 * c + 16 * half);
      s0 = wmma_bf16(aQ[c], b0, s0);
      v16bf b1 = *(const v16bf*)(kt + (size_t)(16 + nl) * DD + 32 * c + 16 * half);
      s1 = wmma_bf16(aQ[c], b1, s1);
    }
#pragma unroll
    for (int j = 0; j < 8; ++j) {
      const int qrow = qb + j + 8 * half;
      float e0 = s0[j] * scale;
      float e1 = s1[j] * scale;
      e0 = (k0 + nl > qrow)      ? -1e30f : e0;   // causal mask
      e1 = (k0 + 16 + nl > qrow) ? -1e30f : e1;
      float mt = rowmax16(fmaxf(e0, e1));
      float mn = fmaxf(m[j], mt);
      float alpha = __expf(m[j] - mn);
      float p0 = __expf(e0 - mn);
      float p1 = __expf(e1 - mn);
      float rs = rowsum16(p0 + p1);
      l[j] = l[j] * alpha + rs;
      m[j] = mn;
#pragma unroll
      for (int t = 0; t < 8; ++t) acc[t][j] *= alpha;
      plds[(j + 8 * half) * 32 + nl]      = (__bf16)p0;
      plds[(j + 8 * half) * 32 + 16 + nl] = (__bf16)p1;
    }
    __syncthreads();                      // single-wave workgroup: cheap
    v16bf aP;
    {
      const __bf16* q = plds + nl * 32 + 16 * half;
      v8bf lo = *(const v8bf*)q;
      v8bf hi = *(const v8bf*)(q + 8);
      aP = cat8(lo, hi);
    }
    __syncthreads();
#pragma unroll
    for (int t = 0; t < 8; ++t) {
      v16bf bv = load_b16(V + (size_t)(16 * t + nl) * TT + k0 + 16 * half);
      acc[t] = wmma_bf16(aP, bv, acc[t]);
    }
  }
  // Normalize and store y in [B, T, H*D] bf16 for the final projection GEMM.
#pragma unroll
  for (int t = 0; t < 8; ++t) {
#pragma unroll
    for (int j = 0; j < 8; ++j) {
      const int qrow = qb + j + 8 * half;
      const int dcol = 16 * t + nl;
      float v = acc[t][j] / l[j];
      yb[((size_t)bb * TT + qrow) * CC + h * DD + dcol] = (__bf16)v;
    }
  }
}

// ---------------------------------------------------------------------------
// Output projection: y(bf16 [B*T, C]) x w_projT(bf16 [C, C]) -> out fp32.
// 32(M) x 64(N) per wave, 8 waves / block.
// ---------------------------------------------------------------------------
__global__ __launch_bounds__(256) void gemm_proj(
    const __bf16* __restrict__ yb, const __bf16* __restrict__ wpT,
    float* __restrict__ out) {
  const int lane = threadIdx.x & 31;
  const int wid  = threadIdx.x >> 5;
  const int half = lane >> 4;
  const int nl   = lane & 15;
  const int task = blockIdx.x * 8 + wid;      // 128 * 32 tasks
  const int mt = task >> 5;
  const int ns = task & 31;
  const int m0 = mt * 32;
  const int n0 = ns * 64;

  v8f acc[2][4] = {};
  for (int k0 = 0; k0 < CC; k0 += 32) {
    v16bf a0 = load_a(yb, m0 + nl,      k0, CC, half);
    v16bf a1 = load_a(yb, m0 + 16 + nl, k0, CC, half);
#pragma unroll
    for (int t = 0; t < 4; ++t) {
      v16bf bf = load_b16(wpT + (size_t)(n0 + 16 * t + nl) * CC + k0 + 16 * half);
      acc[0][t] = wmma_bf16(a0, bf, acc[0][t]);
      acc[1][t] = wmma_bf16(a1, bf, acc[1][t]);
    }
  }
#pragma unroll
  for (int sub = 0; sub < 2; ++sub) {
#pragma unroll
    for (int t = 0; t < 4; ++t) {
#pragma unroll
      for (int j = 0; j < 8; ++j) {
        const int g = m0 + sub * 16 + j + 8 * half;
        const int c = n0 + 16 * t + nl;
        out[(size_t)g * CC + c] = acc[sub][t][j];
      }
    }
  }
}

// ---------------------------------------------------------------------------
// Host launcher
// ---------------------------------------------------------------------------
extern "C" void kernel_launch(void* const* d_in, const int* in_sizes, int n_in,
                              void* d_out, int out_size, void* d_ws,
                              size_t ws_size, hipStream_t stream) {
  const float* x      = (const float*)d_in[0];
  const float* w_attn = (const float*)d_in[1];
  const float* w_proj = (const float*)d_in[2];
  const float* freqs  = (const float*)d_in[3];
  const float* delta  = (const float*)d_in[4];
  float* out = (float*)d_out;

  // Workspace layout (bytes), ~112 MB total:
  char* ws = (char*)d_ws;
  __bf16* xb  = (__bf16*)(ws + 0);           //  16 MB  x in bf16 [B*T, C]
  __bf16* waT = (__bf16*)(ws + 16777216);    //  24 MB  w_attn^T [3C, C]
  __bf16* wpT = (__bf16*)(ws + 41943040);    //   8 MB  w_proj^T [C, C]
  __bf16* qw  = (__bf16*)(ws + 50331648);    //  16 MB  Q [B,H,T,D]
  __bf16* kw  = (__bf16*)(ws + 67108864);    //  16 MB  K [B,H,T,D]
  __bf16* vt  = (__bf16*)(ws + 83886080);    //  16 MB  V^T [B,H,D,T]
  __bf16* yb  = (__bf16*)(ws + 100663296);   //  16 MB  y [B,T,C]

  cvt_bf16<<<(BB * TT * CC) / 256, 256, 0, stream>>>(x, xb, BB * TT * CC);
  cvt_T<<<(CC * 3 * CC) / 256, 256, 0, stream>>>(w_attn, waT, CC, 3 * CC);
  cvt_T<<<(CC * CC) / 256, 256, 0, stream>>>(w_proj, wpT, CC, CC);

  gemm_qkv<<<(128 * 96) / 8, 256, 0, stream>>>(xb, waT, qw, kw, vt);
  rope_kernel<<<(2 * BB * HH * TT * MROT) / 256, 256, 0, stream>>>(qw, kw,
                                                                   freqs, delta);
  attn_kernel<<<BB * HH * (TT / 16), 32, 0, stream>>>(qw, kw, vt, yb);
  gemm_proj<<<(128 * 32) / 8, 256, 0, stream>>>(yb, wpT, out);
}
